// Detector_22462678958492
// MI455X (gfx1250) — compile-verified, matrix operand order checked by reading, not run on hardware
//
#include <hip/hip_runtime.h>
#include <hip/hip_bf16.h>

typedef __attribute__((ext_vector_type(16))) __bf16 v16bf;
typedef __attribute__((ext_vector_type(8)))  float  v8f;

// ---------------------------------------------------------------------------
// WMMA fragment index helper (CDNA5 16-bit A-matrix 16x32 layout, wave32).
// lane L (hi = L>=16): element i of v16bf -> VGPR i/2, half i%2.
// K(v, half) = (v<4 ? 2v : 16+2(v-4)) + 8*hi + half
// B mirrors A with N in place of M; staged in LDS as [n][k].
// ---------------------------------------------------------------------------
__device__ __forceinline__ int frag_k(int i, int hi) {
    int v = i >> 1;
    int base = (v < 4) ? (2 * v) : (16 + 2 * (v - 4));
    return base + hi * 8 + (i & 1);
}

// ---------------------------------------------------------------------------
// Implicit-GEMM 3x3 SAME conv + bias + ReLU, NHWC fp32 in/out, bf16 WMMA core.
// Block = 256 threads = 8 wave32 waves. Block tile: 64 pixels x 128 Cout.
// Each wave: 4 M-subtiles x 16 Cout, B fragment reused across 4 WMMAs.
// ---------------------------------------------------------------------------
__global__ __launch_bounds__(256)
void conv3x3_wmma(const float* __restrict__ in, const float* __restrict__ wgt,
                  const float* __restrict__ bias, float* __restrict__ out,
                  int H, int W, int Cin, int Cout)
{
    __shared__ __bf16 lA[64 * 32];     // [m][k]
    __shared__ __bf16 lB[128 * 32];    // [n][k] (transposed while loading)

    const int tid  = threadIdx.x;
    const int lane = tid & 31;
    const int wave = tid >> 5;
    const int P    = H * W;
    const int Ktot = 9 * Cin;
    const int p0   = blockIdx.x * 64;          // pixel tile base
    const int n0b  = blockIdx.y * 128;         // Cout block base
    const int nw   = n0b + wave * 16;          // this wave's Cout base
    const bool waveActive = (nw < Cout);       // wave-uniform: EXEC stays full

    v8f acc[4] = {};

    for (int k0 = 0; k0 < Ktot; k0 += 32) {
        // ---- stage A tile (64 pixels x 32 K) as bf16 ----
        #pragma unroll
        for (int t = 0; t < 8; ++t) {
            int idx = t * 256 + tid;
            int m = idx >> 5, kk = idx & 31;
            int K = k0 + kk;
            int p = p0 + m;
            float v = 0.f;
            if (K < Ktot && p < P) {
                int y   = p / W, x = p - y * W;
                int ky  = K / (3 * Cin);
                int rem = K - ky * (3 * Cin);
                int kx  = rem / Cin;
                int ci  = rem - kx * Cin;
                int iy = y + ky - 1, ix = x + kx - 1;
                if (iy >= 0 && iy < H && ix >= 0 && ix < W)
                    v = in[(iy * W + ix) * Cin + ci];
            }
            lA[idx] = (__bf16)v;
        }
        // ---- stage B tile (128 Cout x 32 K), transposed, coalesced on Cout ----
        #pragma unroll
        for (int t = 0; t < 16; ++t) {
            int idx = t * 256 + tid;
            int kk = idx >> 7, nn = idx & 127;
            int K = k0 + kk, co = n0b + nn;
            float v = (K < Ktot && co < Cout) ? wgt[K * Cout + co] : 0.f;
            lB[nn * 32 + kk] = (__bf16)v;
        }
        __syncthreads();

        if (waveActive) {
            const int col = lane & 15;
            const int hi  = lane >> 4;
            v16bf b;
            #pragma unroll
            for (int i = 0; i < 16; ++i)
                b[i] = lB[(wave * 16 + col) * 32 + frag_k(i, hi)];
            #pragma unroll
            for (int s = 0; s < 4; ++s) {
                v16bf a;
                #pragma unroll
                for (int i = 0; i < 16; ++i)
                    a[i] = lA[(s * 16 + col) * 32 + frag_k(i, hi)];
                acc[s] = __builtin_amdgcn_wmma_f32_16x16x32_bf16(
                    false, a, false, b, (short)0, acc[s], false, false);
            }
        }
        __syncthreads();
    }

    // ---- store D (C/D layout: vgpr r -> M = r + 8*hi, N = lane&15) ----
    if (waveActive) {
        int hi = lane >> 4;
        int co = nw + (lane & 15);
        if (co < Cout) {
            float bs = bias[co];
            #pragma unroll
            for (int s = 0; s < 4; ++s) {
                #pragma unroll
                for (int r = 0; r < 8; ++r) {
                    int p = p0 + s * 16 + r + hi * 8;
                    if (p < P) {
                        float v = acc[s][r] + bs;
                        out[p * Cout + co] = v > 0.f ? v : 0.f;
                    }
                }
            }
        }
    }
}

// ---------------------------------------------------------------------------
// K-split partial GEMM: C[M,N] += A[M,K(kbeg:kend)] @ B[kbeg:kend, N].
// Block tile: 64 rows x 128 cols (covers all M=64 ROIs). Partial sums are
// committed with global_atomic_add_f32; C must be pre-zeroed.
// Grid (N/128, K-chunks) => 256 blocks to saturate 23.3 TB/s on W_fc.
// ---------------------------------------------------------------------------
__global__ __launch_bounds__(256)
void gemm_wmma_part(const float* __restrict__ A, const float* __restrict__ B,
                    float* __restrict__ C, int M, int N, int K, int kchunk)
{
    __shared__ __bf16 lA[64 * 32];
    __shared__ __bf16 lB[128 * 32];

    const int tid  = threadIdx.x;
    const int lane = tid & 31;
    const int wave = tid >> 5;
    const int n0b  = blockIdx.x * 128;
    const int kbeg = blockIdx.y * kchunk;
    int kend = kbeg + kchunk;
    if (kend > K) kend = K;
    const int nw = n0b + wave * 16;
    const bool waveActive = (nw < N);

    v8f acc[4] = {};

    for (int k0 = kbeg; k0 < kend; k0 += 32) {
        if (k0 + 32 < kend)   // stream next K-slice of B toward L2
            __builtin_prefetch(&B[(size_t)(k0 + 32) * N + n0b + tid], 0, 1);

        #pragma unroll
        for (int t = 0; t < 8; ++t) {
            int idx = t * 256 + tid;
            int m = idx >> 5, kk = idx & 31;
            int Kk = k0 + kk;
            lA[idx] = (__bf16)((m < M && Kk < kend) ? A[(size_t)m * K + Kk] : 0.f);
        }
        #pragma unroll
        for (int t = 0; t < 16; ++t) {
            int idx = t * 256 + tid;
            int kk = idx >> 7, nn = idx & 127;
            int Kk = k0 + kk, co = n0b + nn;
            lB[nn * 32 + kk] =
                (__bf16)((Kk < kend && co < N) ? B[(size_t)Kk * N + co] : 0.f);
        }
        __syncthreads();

        if (waveActive) {
            const int col = lane & 15;
            const int hi  = lane >> 4;
            v16bf b;
            #pragma unroll
            for (int i = 0; i < 16; ++i)
                b[i] = lB[(wave * 16 + col) * 32 + frag_k(i, hi)];
            #pragma unroll
            for (int s = 0; s < 4; ++s) {
                v16bf a;
                #pragma unroll
                for (int i = 0; i < 16; ++i)
                    a[i] = lA[(s * 16 + col) * 32 + frag_k(i, hi)];
                acc[s] = __builtin_amdgcn_wmma_f32_16x16x32_bf16(
                    false, a, false, b, (short)0, acc[s], false, false);
            }
        }
        __syncthreads();
    }

    if (waveActive) {
        int hi = lane >> 4;
        int co = nw + (lane & 15);
        if (co < N) {
            #pragma unroll
            for (int s = 0; s < 4; ++s) {
                #pragma unroll
                for (int r = 0; r < 8; ++r) {
                    int row = s * 16 + r + hi * 8;
                    if (row < M)
                        atomicAdd(&C[(size_t)row * N + co], acc[s][r]);
                }
            }
        }
    }
}

// ---------------------------------------------------------------------------
// Epilogue for the K-split GEMM: C = relu(C + bias).
// ---------------------------------------------------------------------------
__global__ void bias_relu_kernel(float* __restrict__ C, const float* __restrict__ bias,
                                 int M, int N)
{
    int idx = blockIdx.x * blockDim.x + threadIdx.x;
    if (idx < M * N) {
        float v = C[idx] + bias[idx % N];
        C[idx] = v > 0.f ? v : 0.f;
    }
}

// ---------------------------------------------------------------------------
// 2x2 stride-2 maxpool, NHWC.
// ---------------------------------------------------------------------------
__global__ void maxpool2x2(const float* __restrict__ in, float* __restrict__ out,
                           int Ho, int Wo, int C, int Win)
{
    int idx = blockIdx.x * blockDim.x + threadIdx.x;
    int total = Ho * Wo * C;
    if (idx >= total) return;
    int c = idx % C;
    int t = idx / C;
    int xo = t % Wo, yo = t / Wo;
    const float* p = in + ((yo * 2) * Win + xo * 2) * C + c;
    float a = p[0], b = p[C], d = p[Win * C], e = p[Win * C + C];
    out[idx] = fmaxf(fmaxf(a, b), fmaxf(d, e));
}

// ---------------------------------------------------------------------------
// ROI pool: crop on stride-16 grid + bilinear resize to 7x7 (half-pixel).
// One block per (roi, py, px); threads stride channels.
// ---------------------------------------------------------------------------
__global__ void roi_pool_kernel(const float* __restrict__ feat,
                                const float* __restrict__ rois,
                                float* __restrict__ pooled,
                                int Hf, int Wf, int C)
{
    int r  = blockIdx.y;
    int py = blockIdx.x / 7;
    int px = blockIdx.x % 7;

    float cx = rois[r * 4 + 0], cy = rois[r * 4 + 1];
    float w  = rois[r * 4 + 2], h  = rois[r * 4 + 3];
    float x0 = rintf((cx - 0.5f * w) / 16.f);
    float y0 = rintf((cy - 0.5f * h) / 16.f);
    float wq = fmaxf(rintf(w / 16.f), 1.f);
    float hq = fmaxf(rintf(h / 16.f), 1.f);

    float ty = ((float)py + 0.5f) * hq / 7.f - 0.5f;
    ty = fminf(fmaxf(ty, 0.f), hq - 1.f) + y0;
    ty = fminf(fmaxf(ty, 0.f), (float)Hf - 1.f);
    int ya = (int)floorf(ty);
    float fy = ty - (float)ya;
    int yb = min(ya + 1, Hf - 1);

    float tx = ((float)px + 0.5f) * wq / 7.f - 0.5f;
    tx = fminf(fmaxf(tx, 0.f), wq - 1.f) + x0;
    tx = fminf(fmaxf(tx, 0.f), (float)Wf - 1.f);
    int xa = (int)floorf(tx);
    float fx = tx - (float)xa;
    int xb = min(xa + 1, Wf - 1);

    const float* paa = feat + (ya * Wf + xa) * C;
    const float* pab = feat + (ya * Wf + xb) * C;
    const float* pba = feat + (yb * Wf + xa) * C;
    const float* pbb = feat + (yb * Wf + xb) * C;
    float* op = pooled + (((size_t)r * 7 + py) * 7 + px) * C;

    for (int c = threadIdx.x; c < C; c += blockDim.x) {
        float top = paa[c] * (1.f - fx) + pab[c] * fx;
        float bot = pba[c] * (1.f - fx) + pbb[c] * fx;
        op[c] = top * (1.f - fy) + bot * fy;
    }
}

// ---------------------------------------------------------------------------
// Heads: cls logits (21) + softmax, reg (84). One block per ROI.
// ---------------------------------------------------------------------------
__global__ __launch_bounds__(128)
void head_kernel(const float* __restrict__ hfc,
                 const float* __restrict__ Wcls, const float* __restrict__ bcls,
                 const float* __restrict__ Wreg, const float* __restrict__ breg,
                 float* __restrict__ out_cls, float* __restrict__ out_reg)
{
    __shared__ float logit[21];
    int r = blockIdx.x;
    int t = threadIdx.x;
    const float* hr = hfc + (size_t)r * 4096;

    if (t < 21) {
        float s = bcls[t];
        for (int k = 0; k < 4096; ++k) s += hr[k] * Wcls[k * 21 + t];
        logit[t] = s;
    } else if (t < 21 + 84) {
        int j = t - 21;
        float s = breg[j];
        for (int k = 0; k < 4096; ++k) s += hr[k] * Wreg[k * 84 + j];
        out_reg[r * 84 + j] = s;
    }
    __syncthreads();
    if (t < 21) {
        float mx = -1e30f;
        for (int i = 0; i < 21; ++i) mx = fmaxf(mx, logit[i]);
        float sum = 0.f;
        for (int i = 0; i < 21; ++i) sum += expf(logit[i] - mx);
        out_cls[r * 21 + t] = expf(logit[t] - mx) / sum;
    }
}

// ---------------------------------------------------------------------------
// Host orchestration.
// Inputs (dict-leaf order): [0] image, [1..26] (w_i, b_i) x13,
// [27] W_fc, [28] b_fc, [29] W_cls, [30] b_cls, [31] W_reg, [32] b_reg, [33] rois
// ---------------------------------------------------------------------------
extern "C" void kernel_launch(void* const* d_in, const int* in_sizes, int n_in,
                              void* d_out, int out_size, void* d_ws, size_t ws_size,
                              hipStream_t stream)
{
    const float* image = (const float*)d_in[0];
    const float* Wfc  = (const float*)d_in[27];
    const float* bfc  = (const float*)d_in[28];
    const float* Wcls = (const float*)d_in[29];
    const float* bcls = (const float*)d_in[30];
    const float* Wreg = (const float*)d_in[31];
    const float* breg = (const float*)d_in[32];
    const float* rois = (const float*)d_in[33];

    // Workspace layout (bytes):
    //   bufA   @ 0            : 600*800*64 fp32 = 122,880,000
    //   bufB   @ 122,880,000  : 122,880,000
    //   pooled @ 245,760,000  : 64*7*7*512 fp32 = 6,422,528
    //   hfc    @ 252,182,528  : 64*4096 fp32    = 1,048,576
    char* ws = (char*)d_ws;
    float* bufA   = (float*)(ws);
    float* bufB   = (float*)(ws + 122880000ull);
    float* pooled = (float*)(ws + 245760000ull);
    float* hfc    = (float*)(ws + 252182528ull);

    static const int cfg_cin[13]  = {3, 64, 64, 128, 128, 256, 256, 256, 512, 512, 512, 512, 512};
    static const int cfg_cout[13] = {64, 64, 128, 128, 256, 256, 256, 512, 512, 512, 512, 512, 512};

    int H = 600, W = 800;
    const float* cur = image;

    for (int i = 0; i < 13; ++i) {
        int cin = cfg_cin[i], cout = cfg_cout[i];
        float* dst = (cur == (const float*)bufA) ? bufB : bufA;
        dim3 grid((H * W + 63) / 64, (cout + 127) / 128);
        conv3x3_wmma<<<grid, 256, 0, stream>>>(
            cur, (const float*)d_in[1 + 2 * i], (const float*)d_in[2 + 2 * i],
            dst, H, W, cin, cout);
        cur = dst;
        if (i == 1 || i == 3 || i == 6 || i == 9) {
            float* d2 = (cur == (const float*)bufA) ? bufB : bufA;
            int Ho = H / 2, Wo = W / 2;
            int total = Ho * Wo * cout;
            maxpool2x2<<<(total + 255) / 256, 256, 0, stream>>>(cur, d2, Ho, Wo, cout, W);
            cur = d2;
            H = Ho; W = Wo;
        }
    }

    // cur: final feature map [37][50][512]
    dim3 rg(49, 64);
    roi_pool_kernel<<<rg, 256, 0, stream>>>(cur, rois, pooled, H, W, 512);

    // FC: hfc = relu(pooled[64,25088] @ W_fc[25088,4096] + b_fc)
    // K split into 8 chunks of 3136 (98 WMMA steps) -> 32x8 = 256 blocks.
    hipMemsetAsync(hfc, 0, 64 * 4096 * sizeof(float), stream);
    dim3 fg(4096 / 128, 8);
    gemm_wmma_part<<<fg, 256, 0, stream>>>(pooled, Wfc, hfc, 64, 4096, 25088, 3136);
    bias_relu_kernel<<<(64 * 4096 + 255) / 256, 256, 0, stream>>>(hfc, bfc, 64, 4096);

    float* out_cls = (float*)d_out;          // [64,21]
    float* out_reg = out_cls + 64 * 21;      // [64,84]
    head_kernel<<<64, 128, 0, stream>>>(hfc, Wcls, bcls, Wreg, breg, out_cls, out_reg);
}
